// Network_21861383536827
// MI455X (gfx1250) — matrix operand order repaired and checked by reading, not compiled, not run on hardware
//
#include <hip/hip_runtime.h>
#include <hip/hip_bf16.h>
#include <math.h>

typedef _Float16 half_t;
typedef _Float16 half8 __attribute__((ext_vector_type(8)));
typedef _Float16 v16h  __attribute__((ext_vector_type(16)));
typedef float    v8f   __attribute__((ext_vector_type(8)));

#define LN_EPS 1e-5f

// ---------------- device helpers ----------------

__device__ __forceinline__ v16h hcomb(half8 lo, half8 hi) {
  union { v16h v; half8 h[2]; } u;
  u.h[0] = lo; u.h[1] = hi;
  return u.v;
}

// load 8 consecutive f32 and convert to 8 halfs
__device__ __forceinline__ half8 ld_f32x8_h8(const float* __restrict__ p) {
  float4 a = *(const float4*)p;
  float4 b = *(const float4*)(p + 4);
  half8 r;
  r[0] = (half_t)a.x; r[1] = (half_t)a.y; r[2] = (half_t)a.z; r[3] = (half_t)a.w;
  r[4] = (half_t)b.x; r[5] = (half_t)b.y; r[6] = (half_t)b.z; r[7] = (half_t)b.w;
  return r;
}

__device__ __forceinline__ v8f wmma_fp16(v16h a, v16h b, v8f c) {
  // D = A(16x32 f16) * B(32x16 f16) + C(16x16 f32)
  return __builtin_amdgcn_wmma_f32_16x16x32_f16(false, a, false, b, (short)0, c,
                                                false, false);
}

// DPP16 ROW_XMASK butterfly step: pure VALU cross-lane within a 16-lane row
// (no ds_bpermute / LDS traffic). dpp_ctrl = 0x160 | mask.
template <int MASK>
__device__ __forceinline__ float xor16_add(float x) {
  int y = __builtin_amdgcn_update_dpp(0, __float_as_int(x), 0x160 | MASK, 0xf, 0xf, true);
  return x + __int_as_float(y);
}
template <int MASK>
__device__ __forceinline__ float xor16_max(float x) {
  int y = __builtin_amdgcn_update_dpp(0, __float_as_int(x), 0x160 | MASK, 0xf, 0xf, true);
  return fmaxf(x, __int_as_float(y));
}
__device__ __forceinline__ float rowsum16(float x) {
  x = xor16_add<1>(x); x = xor16_add<2>(x);
  x = xor16_add<4>(x); x = xor16_add<8>(x);
  return x;
}
__device__ __forceinline__ float rowmax16(float x) {
  x = xor16_max<1>(x); x = xor16_max<2>(x);
  x = xor16_max<4>(x); x = xor16_max<8>(x);
  return x;
}

// Fused bias + ReLU + LayerNorm(16) epilogue on a 16x16 f32 D fragment.
// D layout: VGPR r, lanes 0-15 -> row r, lanes 16-31 -> row r+8; col = lane&15.
// Each matrix row lives in ONE VGPR across one 16-lane DPP row, so ROW_XMASK
// butterflies reduce exactly over the LayerNorm axis.
template <bool OUT_HALF>
__device__ __forceinline__ void dense_epilogue(v8f acc, const float* __restrict__ bias,
                                               const float* __restrict__ gamma,
                                               const float* __restrict__ beta,
                                               void* __restrict__ dst, long row0, int lane) {
  int col = lane & 15, hi = lane >> 4;
  float bc = bias[col], gc = gamma[col], btc = beta[col];
#pragma unroll
  for (int r = 0; r < 8; ++r) {
    float h = acc[r] + bc;
    h = h > 0.f ? h : 0.f;                       // ReLU
    float mu = rowsum16(h) * 0.0625f;
    float d = h - mu;
    float var = rowsum16(d * d) * 0.0625f;
    float o = d * rsqrtf(var + LN_EPS) * gc + btc;
    long row = row0 + r + 8 * hi;
    if (OUT_HALF) ((half_t*)dst)[row * 16 + col] = (half_t)o;
    else          ((float*)dst)[row * 16 + col]  = o;
  }
}

// float <-> monotonic unsigned for atomicMax on floats
__device__ __forceinline__ unsigned flipF(float f) {
  unsigned u = __float_as_uint(f);
  return (u & 0x80000000u) ? ~u : (u | 0x80000000u);
}
__device__ __forceinline__ float unflipF(unsigned u) {
  u = (u & 0x80000000u) ? (u & 0x7fffffffu) : ~u;
  return __uint_as_float(u);
}

// ---------------- kernels ----------------

// Pack W (din x 16, f32 row-major, zero-padded to nchunks*32 in K) into the
// B-fragment layout: Bp[(c*32 + lane)*16 + j] = W[(c*32 + (j<8 ? 8*(lane>>4) + j : 16 + 8*(lane>>4) + j - 8))*16 + (lane&15)]
__global__ void pack_b_k(const float* __restrict__ W, int din, int nchunks,
                         half_t* __restrict__ out) {
  int tid = blockIdx.x * blockDim.x + threadIdx.x;
  int total = nchunks * 32 * 16;
  if (tid >= total) return;
  int j = tid & 15;
  int l = (tid >> 4) & 31;
  int c = tid >> 9;
  int K = c * 32 + ((j < 8) ? ((l >> 4) * 8 + j) : (16 + (l >> 4) * 8 + (j - 8)));
  int n = l & 15;
  out[tid] = (K < din) ? (half_t)W[K * 16 + n] : (half_t)0.f;
}

__global__ void f2h_k(const float* __restrict__ in, half_t* __restrict__ out, long n) {
  long i = (long)blockIdx.x * blockDim.x + threadIdx.x;
  if (i < n) out[i] = (half_t)in[i];
}

// out[row, 0:16] = lat[row], out[row, 16:32] = dat[row]  (all f16)
__global__ void concat_k(const half_t* __restrict__ lat, const half_t* __restrict__ dat,
                         long rows, half_t* __restrict__ out) {
  long i = (long)blockIdx.x * blockDim.x + threadIdx.x;
  if (i >= rows * 32) return;
  long row = i >> 5; int c = (int)(i & 31);
  out[i] = (c < 16) ? lat[row * 16 + c] : dat[row * 16 + (c - 16)];
}

// Dense block, din=16 (padded K to 32): src f32 [rows,16] -> dst [rows,16]
template <bool OUT_HALF>
__global__ void dense16_k(const float* __restrict__ src, long rows,
                          const half_t* __restrict__ Bp, const float* __restrict__ bias,
                          const float* __restrict__ gamma, const float* __restrict__ beta,
                          void* __restrict__ dst) {
  int lane = threadIdx.x & 31;
  long tile = (long)blockIdx.x * (blockDim.x >> 5) + (threadIdx.x >> 5);
  if (tile * 16 >= rows) return;
  int hi = lane >> 4, r = lane & 15;
  const float* rowp = src + (tile * 16 + r) * 16;
  half8 z = {};
  v16h a = hcomb(ld_f32x8_h8(rowp + 8 * hi), z);    // K 0..15 real, 16..31 pad
  v16h b = *(const v16h*)(Bp + lane * 16);
  v8f acc = {};
  acc = wmma_fp16(a, b, acc);
  dense_epilogue<OUT_HALF>(acc, bias, gamma, beta, dst, tile * 16, lane);
}

// Edge core: e_in = [ce | cx[src] | cx[dst] | cg[eg]] (128) -> e2 [E,16] f32.
// Each 32-wide K chunk == one concat segment, so the A fragment is two
// contiguous b128 loads from the right segment row.
__global__ void core_edge_k(const half_t* __restrict__ ce, const half_t* __restrict__ cx,
                            const half_t* __restrict__ cg, const int* __restrict__ edges,
                            const int* __restrict__ eg_idx, long E,
                            const half_t* __restrict__ Bp, const float* __restrict__ bias,
                            const float* __restrict__ gamma, const float* __restrict__ beta,
                            float* __restrict__ e2) {
  int lane = threadIdx.x & 31;
  long tile = (long)blockIdx.x * (blockDim.x >> 5) + (threadIdx.x >> 5);
  if (tile * 16 >= E) return;
  int hi = lane >> 4, r = lane & 15;
  long e = tile * 16 + r;
  int si = edges[e], di = edges[E + e], gi = eg_idx[e];
  const half_t* segs[4] = { ce + e * 32, cx + (long)si * 32,
                            cx + (long)di * 32, cg + (long)gi * 32 };
  v8f acc = {};
#pragma unroll
  for (int c = 0; c < 4; ++c) {
    const half_t* p = segs[c];
    v16h a = hcomb(*(const half8*)(p + 8 * hi), *(const half8*)(p + 16 + 8 * hi));
    v16h b = *(const v16h*)(Bp + (c * 32 + lane) * 16);
    acc = wmma_fp16(a, b, acc);
  }
  dense_epilogue<false>(acc, bias, gamma, beta, e2, tile * 16, lane);
}

// Node core: x_in = [cx(32) | agg_e(16 f32) | cg[node_idx](32)] pad->96 -> x2 [N,16] f32
__global__ void core_node_k(const half_t* __restrict__ cx, const float* __restrict__ agg_e,
                            const half_t* __restrict__ cg, const int* __restrict__ node_idx,
                            long N, const half_t* __restrict__ Bp,
                            const float* __restrict__ bias, const float* __restrict__ gamma,
                            const float* __restrict__ beta, float* __restrict__ x2) {
  int lane = threadIdx.x & 31;
  long tile = (long)blockIdx.x * (blockDim.x >> 5) + (threadIdx.x >> 5);
  if (tile * 16 >= N) return;
  int hi = lane >> 4, r = lane & 15;
  long n = tile * 16 + r;
  int gi = node_idx[n];
  const half_t* px = cx + n * 32;
  const float*  pa = agg_e + n * 16;
  const half_t* pg = cg + (long)gi * 32;
  v8f acc = {};
  { // chunk 0: K 0..31 from cx
    v16h a = hcomb(*(const half8*)(px + 8 * hi), *(const half8*)(px + 16 + 8 * hi));
    acc = wmma_fp16(a, *(const v16h*)(Bp + (0 * 32 + lane) * 16), acc);
  }
  { // chunk 1: K 32..47 agg_e (f32), K 48..63 cg[0..15]
    v16h a = hcomb(ld_f32x8_h8(pa + 8 * hi), *(const half8*)(pg + 8 * hi));
    acc = wmma_fp16(a, *(const v16h*)(Bp + (1 * 32 + lane) * 16), acc);
  }
  { // chunk 2: K 64..79 cg[16..31], K 80..95 pad
    half8 z = {};
    v16h a = hcomb(*(const half8*)(pg + 16 + 8 * hi), z);
    acc = wmma_fp16(a, *(const v16h*)(Bp + (2 * 32 + lane) * 16), acc);
  }
  dense_epilogue<false>(acc, bias, gamma, beta, x2, tile * 16, lane);
}

// Graph core: g_in = [cg(32) | agg_eg(16 f32) | agg_xg(16 f32)] -> g2 [G,16] f32
__global__ void core_graph_k(const half_t* __restrict__ cg, const float* __restrict__ agg_eg,
                             const float* __restrict__ agg_xg, long G,
                             const half_t* __restrict__ Bp, const float* __restrict__ bias,
                             const float* __restrict__ gamma, const float* __restrict__ beta,
                             float* __restrict__ g2) {
  int lane = threadIdx.x & 31;
  long tile = (long)blockIdx.x * (blockDim.x >> 5) + (threadIdx.x >> 5);
  if (tile * 16 >= G) return;
  int hi = lane >> 4, r = lane & 15;
  long g = tile * 16 + r;
  const half_t* pg = cg + g * 32;
  const float*  pe = agg_eg + g * 16;
  const float*  px = agg_xg + g * 16;
  v8f acc = {};
  { v16h a = hcomb(*(const half8*)(pg + 8 * hi), *(const half8*)(pg + 16 + 8 * hi));
    acc = wmma_fp16(a, *(const v16h*)(Bp + (0 * 32 + lane) * 16), acc); }
  { v16h a = hcomb(ld_f32x8_h8(pe + 8 * hi), ld_f32x8_h8(px + 8 * hi));
    acc = wmma_fp16(a, *(const v16h*)(Bp + (1 * 32 + lane) * 16), acc); }
  dense_epilogue<false>(acc, bias, gamma, beta, g2, tile * 16, lane);
}

// segment sums: e2 scattered to agg_e (by dst) and agg_eg (by edge graph)
__global__ void scatter_e_k(const float* __restrict__ e2, const int* __restrict__ edges,
                            const int* __restrict__ eg_idx, long E,
                            float* __restrict__ agg_e, float* __restrict__ agg_eg) {
  long i = (long)blockIdx.x * blockDim.x + threadIdx.x;
  if (i >= E * 16) return;
  long e = i >> 4; int c = (int)(i & 15);
  float v = e2[i];
  atomicAdd(&agg_e[(long)edges[E + e] * 16 + c], v);
  atomicAdd(&agg_eg[(long)eg_idx[e] * 16 + c], v);
}

__global__ void scatter_x_k(const float* __restrict__ x2, const int* __restrict__ node_idx,
                            long N, float* __restrict__ agg_xg) {
  long i = (long)blockIdx.x * blockDim.x + threadIdx.x;
  if (i >= N * 16) return;
  long n = i >> 4; int c = (int)(i & 15);
  atomicAdd(&agg_xg[(long)node_idx[n] * 16 + c], x2[i]);
}

// out-layer logits (C<=2) + half-wave-reduced global column max.
// Reduction stays within 16-lane DPP rows; lanes 0 and 16 both issue atomics.
__global__ void logits_k(const half_t* __restrict__ src, long rows,
                         const float* __restrict__ W, const float* __restrict__ b, int C,
                         float* __restrict__ logits, unsigned* __restrict__ maxbuf) {
  long row = (long)blockIdx.x * blockDim.x + threadIdx.x;
  float a0 = -3.0e38f, a1 = -3.0e38f;
  if (row < rows) {
    a0 = b[0]; a1 = (C > 1) ? b[1] : 0.f;
#pragma unroll
    for (int k = 0; k < 16; ++k) {
      float v = (float)src[row * 16 + k];
      a0 += v * W[k * C];
      if (C > 1) a1 += v * W[k * C + 1];
    }
    logits[row * C] = a0;
    if (C > 1) logits[row * C + 1] = a1;
  }
  float m0 = rowmax16(a0);
  float m1 = rowmax16(a1);
  if ((threadIdx.x & 15) == 0) {
    atomicMax(&maxbuf[0], flipF(m0));
    if (C > 1) atomicMax(&maxbuf[1], flipF(m1));
  }
}

__global__ void expsum_k(const float* __restrict__ logits, long rows, int C,
                         const unsigned* __restrict__ maxbuf, float* __restrict__ sumbuf) {
  long row = (long)blockIdx.x * blockDim.x + threadIdx.x;
  float s0 = 0.f, s1 = 0.f;
  if (row < rows) {
    s0 = expf(logits[row * C] - unflipF(maxbuf[0]));
    if (C > 1) s1 = expf(logits[row * C + 1] - unflipF(maxbuf[1]));
  }
  s0 = rowsum16(s0);
  s1 = rowsum16(s1);
  if ((threadIdx.x & 15) == 0) {
    atomicAdd(&sumbuf[0], s0);
    if (C > 1) atomicAdd(&sumbuf[1], s1);
  }
}

__global__ void smwrite_k(const float* __restrict__ logits, long rows, int C,
                          const unsigned* __restrict__ maxbuf, const float* __restrict__ sumbuf,
                          float* __restrict__ out) {
  long row = (long)blockIdx.x * blockDim.x + threadIdx.x;
  if (row >= rows) return;
  for (int c = 0; c < C; ++c)
    out[row * C + c] = expf(logits[row * C + c] - unflipF(maxbuf[c])) / sumbuf[c];
}

// ---------------- host ----------------

extern "C" void kernel_launch(void* const* d_in, const int* in_sizes, int n_in,
                              void* d_out, int out_size, void* d_ws, size_t ws_size,
                              hipStream_t stream) {
  const float* x_f32 = (const float*)d_in[0];
  const float* e_f32 = (const float*)d_in[1];
  const float* g_f32 = (const float*)d_in[2];
  const int* edges    = (const int*)d_in[3];
  const int* node_idx = (const int*)d_in[4];
  const int* edge_idx = (const int*)d_in[5];
  const long N = in_sizes[4];
  const long E = in_sizes[5];
  const long G = in_sizes[2] / 16;
  const int STEPS = 2;

  // param base indices: enc_e=6 enc_n=10 enc_g=14 core_e=18 core_n=22 core_g=26
  auto P = [&](int base, int off) { return (const float*)d_in[base + off]; };

  // workspace bump allocator (256B aligned)
  char* wsb = (char*)d_ws;
  size_t off = 0;
  auto alloc = [&](size_t bytes) -> void* {
    void* p = wsb + off;
    off += (bytes + 255) & ~(size_t)255;
    return p;
  };
  half_t* BpEncE = (half_t*)alloc(1 * 512 * sizeof(half_t));
  half_t* BpEncN = (half_t*)alloc(1 * 512 * sizeof(half_t));
  half_t* BpEncG = (half_t*)alloc(1 * 512 * sizeof(half_t));
  half_t* BpCorE = (half_t*)alloc(4 * 512 * sizeof(half_t));
  half_t* BpCorN = (half_t*)alloc(3 * 512 * sizeof(half_t));
  half_t* BpCorG = (half_t*)alloc(2 * 512 * sizeof(half_t));
  half_t* le = (half_t*)alloc(E * 16 * 2);
  half_t* lx = (half_t*)alloc(N * 16 * 2);
  half_t* lg = (half_t*)alloc(G * 16 * 2);
  half_t* de = (half_t*)alloc(E * 16 * 2);
  half_t* dx = (half_t*)alloc(N * 16 * 2);
  half_t* dg = (half_t*)alloc(G * 16 * 2);
  half_t* ce = (half_t*)alloc(E * 32 * 2);
  half_t* cx = (half_t*)alloc(N * 32 * 2);
  half_t* cg = (half_t*)alloc(G * 32 * 2);
  float* e2 = (float*)alloc(E * 16 * 4);
  float* x2 = (float*)alloc(N * 16 * 4);
  float* g2 = (float*)alloc(G * 16 * 4);
  float* agg_e  = (float*)alloc(N * 16 * 4);
  float* agg_eg = (float*)alloc(G * 16 * 4);
  float* agg_xg = (float*)alloc(G * 16 * 4);
  float* elog = (float*)alloc(E * 2 * 4);
  float* xlog = (float*)alloc(N * 2 * 4);
  float* glog = (float*)alloc(G * 1 * 4);
  void*  red  = alloc(64);
  if (off > ws_size) return;  // not enough scratch; bail safely

  auto cdiv = [](long a, long b) -> int { return (int)((a + b - 1) / b); };
  const int T = 256;  // 8 wave32s per block

  // pack weights into B-fragment layout
  pack_b_k<<<cdiv(1 * 512, T), T, 0, stream>>>(P(6, 0), 16, 1, BpEncE);
  pack_b_k<<<cdiv(1 * 512, T), T, 0, stream>>>(P(10, 0), 16, 1, BpEncN);
  pack_b_k<<<cdiv(1 * 512, T), T, 0, stream>>>(P(14, 0), 16, 1, BpEncG);
  pack_b_k<<<cdiv(4 * 512, T), T, 0, stream>>>(P(18, 0), 128, 4, BpCorE);
  pack_b_k<<<cdiv(3 * 512, T), T, 0, stream>>>(P(22, 0), 80, 3, BpCorN);
  pack_b_k<<<cdiv(2 * 512, T), T, 0, stream>>>(P(26, 0), 64, 2, BpCorG);

  // data0 = f16(inputs)
  f2h_k<<<cdiv(E * 16, T), T, 0, stream>>>(e_f32, de, E * 16);
  f2h_k<<<cdiv(N * 16, T), T, 0, stream>>>(x_f32, dx, N * 16);
  f2h_k<<<cdiv(G * 16, T), T, 0, stream>>>(g_f32, dg, G * 16);

  // latent0 = encode(inputs)
  dense16_k<true><<<cdiv(E / 16, 8), T, 0, stream>>>(e_f32, E, BpEncE, P(6, 1), P(6, 2), P(6, 3), le);
  dense16_k<true><<<cdiv(N / 16, 8), T, 0, stream>>>(x_f32, N, BpEncN, P(10, 1), P(10, 2), P(10, 3), lx);
  dense16_k<true><<<cdiv(G / 16, 8), T, 0, stream>>>(g_f32, G, BpEncG, P(14, 1), P(14, 2), P(14, 3), lg);

  for (int s = 0; s < STEPS; ++s) {
    // concat latent | data
    concat_k<<<cdiv(E * 32, T), T, 0, stream>>>(le, de, E, ce);
    concat_k<<<cdiv(N * 32, T), T, 0, stream>>>(lx, dx, N, cx);
    concat_k<<<cdiv(G * 32, T), T, 0, stream>>>(lg, dg, G, cg);

    hipMemsetAsync(agg_e, 0, N * 16 * 4, stream);
    hipMemsetAsync(agg_eg, 0, G * 16 * 4, stream);
    hipMemsetAsync(agg_xg, 0, G * 16 * 4, stream);

    core_edge_k<<<cdiv(E / 16, 8), T, 0, stream>>>(ce, cx, cg, edges, edge_idx, E,
                                                   BpCorE, P(18, 1), P(18, 2), P(18, 3), e2);
    scatter_e_k<<<cdiv(E * 16, T), T, 0, stream>>>(e2, edges, edge_idx, E, agg_e, agg_eg);
    core_node_k<<<cdiv(N / 16, 8), T, 0, stream>>>(cx, agg_e, cg, node_idx, N,
                                                   BpCorN, P(22, 1), P(22, 2), P(22, 3), x2);
    scatter_x_k<<<cdiv(N * 16, T), T, 0, stream>>>(x2, node_idx, N, agg_xg);
    core_graph_k<<<cdiv(G / 16, 8), T, 0, stream>>>(cg, agg_eg, agg_xg, G,
                                                    BpCorG, P(26, 1), P(26, 2), P(26, 3), g2);

    // encode core outputs -> new data (f16)
    dense16_k<true><<<cdiv(E / 16, 8), T, 0, stream>>>(e2, E, BpEncE, P(6, 1), P(6, 2), P(6, 3), de);
    dense16_k<true><<<cdiv(N / 16, 8), T, 0, stream>>>(x2, N, BpEncN, P(10, 1), P(10, 2), P(10, 3), dx);
    dense16_k<true><<<cdiv(G / 16, 8), T, 0, stream>>>(g2, G, BpEncG, P(14, 1), P(14, 2), P(14, 3), dg);

    // global (axis-0) softmax: logits+max, expsum, normalize
    hipMemsetAsync(red, 0, 64, stream);
    unsigned* maxE = (unsigned*)red;     float* sumE = (float*)red + 2;
    unsigned* maxN = (unsigned*)red + 4; float* sumN = (float*)red + 6;
    unsigned* maxG = (unsigned*)red + 8; float* sumG = (float*)red + 10;

    logits_k<<<cdiv(E, T), T, 0, stream>>>(de, E, (const float*)d_in[30], (const float*)d_in[31], 2, elog, maxE);
    logits_k<<<cdiv(N, T), T, 0, stream>>>(dx, N, (const float*)d_in[32], (const float*)d_in[33], 2, xlog, maxN);
    logits_k<<<cdiv(G, T), T, 0, stream>>>(dg, G, (const float*)d_in[34], (const float*)d_in[35], 1, glog, maxG);

    expsum_k<<<cdiv(E, T), T, 0, stream>>>(elog, E, 2, maxE, sumE);
    expsum_k<<<cdiv(N, T), T, 0, stream>>>(xlog, N, 2, maxN, sumN);
    expsum_k<<<cdiv(G, T), T, 0, stream>>>(glog, G, 1, maxG, sumG);

    float* oe = (float*)d_out + (size_t)s * E * 2;
    float* on = (float*)d_out + (size_t)2 * E * 2 + (size_t)s * N * 2;
    float* og = (float*)d_out + (size_t)2 * E * 2 + (size_t)2 * N * 2 + (size_t)s * G;
    smwrite_k<<<cdiv(E, T), T, 0, stream>>>(elog, E, 2, maxE, sumE, oe);
    smwrite_k<<<cdiv(N, T), T, 0, stream>>>(xlog, N, 2, maxN, sumN, on);
    smwrite_k<<<cdiv(G, T), T, 0, stream>>>(glog, G, 1, maxG, sumG, og);
  }
}
